// HAN_23278722744709
// MI455X (gfx1250) — matrix operand order repaired and checked by reading, not compiled, not run on hardware
//
#include <hip/hip_runtime.h>
#include <hip/hip_bf16.h>

typedef __attribute__((ext_vector_type(16))) _Float16 v16h;
typedef __attribute__((ext_vector_type(8)))  _Float16 v8h;
typedef __attribute__((ext_vector_type(8)))  float    v8f;

#define N_NOTES   16384
#define NUM_BEATS 2048
#define NUM_MEAS  512
#define CMB_PITCH 544   // 523 padded to multiple of 32

// ---------------------------------------------------------------------------
// f32 [rows, scols] (row-major) -> f16 [rows, dpitch], zero-padded columns
// ---------------------------------------------------------------------------
__global__ void k_cvt_f16(const float* __restrict__ src, _Float16* __restrict__ dst,
                          int rows, int scols, int dpitch) {
  int idx = blockIdx.x * blockDim.x + threadIdx.x;
  int total = rows * dpitch;
  if (idx >= total) return;
  int r = idx / dpitch, c = idx - r * dpitch;
  dst[idx] = (c < scols) ? (_Float16)src[(size_t)r * scols + c] : (_Float16)0.f;
}

// ---------------------------------------------------------------------------
// WMMA GEMM:  C[M,N] = A[M,Kp] * W[N,Kp]^T     (f16 in, f32 out)
// One wave per 16x16 tile; 4 waves per block tile 64 rows of M.
// A layout per lane (16-bit A 16x32): M = lane&15, K runs hi*8+[0,8) and 16+hi*8+[0,8)
// B layout per lane (16-bit B 32x16): N = lane&15, K runs hi*8+[0,8) and 16+hi*8+[0,8)
// D layout: v8f element r -> row (hi*8 + r), col (lane&15)
// ---------------------------------------------------------------------------
__global__ void k_gemm_wmma(const _Float16* __restrict__ A,
                            const _Float16* __restrict__ W,
                            float* __restrict__ C,
                            int Kp, int ldc) {
  int lane = threadIdx.x & 31;
  int wave = threadIdx.x >> 5;
  int hi   = lane >> 4;
  int l15  = lane & 15;
  int n0   = blockIdx.x * 16;
  int m0   = blockIdx.y * 64 + wave * 16;

  const _Float16* __restrict__ Ar = A + (size_t)(m0 + l15) * Kp;
  const _Float16* __restrict__ Wr = W + (size_t)(n0 + l15) * Kp;

  v8f acc = {};
  for (int k = 0; k < Kp; k += 32) {
    __builtin_prefetch(Ar + k + 64, 0, 3);
    v8h a0 = *(const v8h*)(Ar + k + hi * 8);
    v8h a1 = *(const v8h*)(Ar + k + 16 + hi * 8);
    v16h a = __builtin_shufflevector(a0, a1, 0,1,2,3,4,5,6,7,8,9,10,11,12,13,14,15);
    v8h b0 = *(const v8h*)(Wr + k + hi * 8);
    v8h b1 = *(const v8h*)(Wr + k + 16 + hi * 8);
    v16h b = __builtin_shufflevector(b0, b1, 0,1,2,3,4,5,6,7,8,9,10,11,12,13,14,15);
    acc = __builtin_amdgcn_wmma_f32_16x16x32_f16(false, a, false, b,
                                                 (short)0, acc, false, false);
  }
  float* Cp = C + (size_t)(m0 + hi * 8) * ldc + n0 + l15;
#pragma unroll
  for (int r = 0; r < 8; ++r) Cp[(size_t)r * ldc] = acc[r];
}

// ---------------------------------------------------------------------------
// Dual-direction serial LSTM recurrence. gridDim.x = 2 for biLSTM (block 0 =
// forward, block 1 = backward chain, each on its own WGP), grid = 1 for the
// unidirectional output LSTM. 4H threads; Whh staged TRANSPOSED in LDS as
// f32 wT[H][4H] so the inner product reads consecutive dwords per lane
// (conflict-free) and hsh[j] is a same-address broadcast. Steady state per
// step: 1 coalesced global load (precomputed gate) + H ds_load + H v_fma.
// torch gate order: i, f, g, o (blocks of H rows).
// ---------------------------------------------------------------------------
__device__ __forceinline__ float sigf(float x) { return 1.f / (1.f + __expf(-x)); }

template<int H>
__global__ void k_lstm_scan2(const float* __restrict__ G0, const float* __restrict__ G1,
                             const float* __restrict__ W0, const float* __restrict__ W1,
                             const float* __restrict__ bi0, const float* __restrict__ bh0,
                             const float* __restrict__ bi1, const float* __restrict__ bh1,
                             float* __restrict__ Hout, int T, int ldh) {
  constexpr int G4 = 4 * H;
  extern __shared__ float smem[];
  float* wT  = smem;            // [H][G4]  transposed Whh
  float* gsh = smem + H * G4;   // [G4] gate preacts
  float* hsh = gsh + G4;        // [H]  hidden state

  int dir = blockIdx.x;
  const float* __restrict__ G   = dir ? G1 : G0;
  const float* __restrict__ Whh = dir ? W1 : W0;
  const float* __restrict__ bi  = dir ? bi1 : bi0;
  const float* __restrict__ bh  = dir ? bh1 : bh0;

  int g = threadIdx.x;          // 0 .. G4-1
  // stage Whh^T: wT[j][g] = Whh[g][j]
  for (int j = 0; j < H; ++j) wT[j * G4 + g] = Whh[(size_t)g * H + j];
  float bias = bi[g] + bh[g];
  float c = 0.f;
  if (g < H) hsh[g] = 0.f;
  __syncthreads();

  for (int s = 0; s < T; ++s) {
    int t = dir ? (T - 1 - s) : s;
    float acc = G[(size_t)t * G4 + g] + bias;
#pragma unroll 16
    for (int j = 0; j < H; ++j) acc = fmaf(wT[j * G4 + g], hsh[j], acc);
    gsh[g] = acc;
    __syncthreads();
    if (g < H) {
      float i  = sigf(gsh[g]);
      float f  = sigf(gsh[H + g]);
      float gg = tanhf(gsh[2 * H + g]);
      float o  = sigf(gsh[3 * H + g]);
      c = f * c + i * gg;
      float h = o * tanhf(c);
      hsh[g] = h;
      Hout[(size_t)t * ldh + dir * H + g] = h;
    }
    __syncthreads();
  }
}

// ---------------------------------------------------------------------------
// Fixed-width segment sum (the "attention" pooling degenerates to a sum).
// ---------------------------------------------------------------------------
__global__ void k_pool_sum(const float* __restrict__ src, float* __restrict__ dst,
                           int segs, int R, int C) {
  int idx = blockIdx.x * blockDim.x + threadIdx.x;
  if (idx >= segs * C) return;
  int s = idx / C, c = idx - s * C;
  float a = 0.f;
  for (int r = 0; r < R; ++r) a += src[(size_t)(s * R + r) * C + c];
  dst[idx] = a;
}

// ---------------------------------------------------------------------------
// Build combined[N,544] (f16): [h_note(256) | h_beat[t/8](128) | h_meas[t/32](128) | y(11) | pad]
// ---------------------------------------------------------------------------
__global__ void k_build_combined(const float* __restrict__ hn,
                                 const float* __restrict__ hb,
                                 const float* __restrict__ hm,
                                 const float* __restrict__ y,
                                 _Float16* __restrict__ dst, int N) {
  int idx = blockIdx.x * blockDim.x + threadIdx.x;
  if (idx >= N * CMB_PITCH) return;
  int t = idx / CMB_PITCH, cidx = idx - t * CMB_PITCH;
  float v = 0.f;
  if      (cidx < 256) v = hn[(size_t)t * 256 + cidx];
  else if (cidx < 384) v = hb[(size_t)(t >> 3) * 128 + (cidx - 256)];
  else if (cidx < 512) v = hm[(size_t)(t >> 5) * 128 + (cidx - 384)];
  else if (cidx < 523) v = y[(size_t)t * 11 + (cidx - 512)];
  dst[idx] = (_Float16)v;
}

// ---------------------------------------------------------------------------
// Final epilogue: notes = h@fc^T+b ; trill = (h@tw^T+tb) * x[:,46], sigmoid last.
// d_out = [N*11 notes | N*5 trill]
// ---------------------------------------------------------------------------
__global__ void k_final(const float* __restrict__ ho,
                        const float* __restrict__ fcw, const float* __restrict__ fcb,
                        const float* __restrict__ tw,  const float* __restrict__ tb,
                        const float* __restrict__ x,
                        float* __restrict__ out, int N) {
  int t = blockIdx.x * blockDim.x + threadIdx.x;
  if (t >= N) return;
  const float* h = ho + (size_t)t * 64;
  float* notes = out + (size_t)t * 11;
  float* trill = out + (size_t)N * 11 + (size_t)t * 5;
  for (int o = 0; o < 11; ++o) {
    float a = fcb[o];
#pragma unroll 8
    for (int j = 0; j < 64; ++j) a += fcw[o * 64 + j] * h[j];
    notes[o] = a;
  }
  float mask = x[(size_t)t * 55 + 46];   // IS_TRILL_INDEX = -9
  for (int o = 0; o < 5; ++o) {
    float a = tb[o];
#pragma unroll 8
    for (int j = 0; j < 64; ++j) a += tw[o * 64 + j] * h[j];
    a *= mask;
    if (o == 4) a = sigf(a);
    trill[o] = a;
  }
}

// ---------------------------------------------------------------------------
extern "C" void kernel_launch(void* const* d_in, const int* in_sizes, int n_in,
                              void* d_out, int out_size, void* d_ws, size_t ws_size,
                              hipStream_t stream) {
  (void)in_sizes; (void)n_in; (void)out_size; (void)ws_size;
  const float* X = (const float*)d_in[0];   // [16384,55]
  const float* Y = (const float*)d_in[1];   // [16384,11]
  const float* FCW = (const float*)d_in[61];
  const float* FCB = (const float*)d_in[62];
  const float* TRW = (const float*)d_in[63];
  const float* TRB = (const float*)d_in[64];

  // LSTM (layer,dir) units: input index of Wih; (Wih,Whh,bih,bhh) consecutive.
  struct Unit { int p; int T; int K; int Kp; int H; };
  static const Unit U[13] = {
    { 5, N_NOTES,  55,  64, 128}, { 9, N_NOTES,  55,  64, 128},   // note L0 f/b
    {13, N_NOTES, 256, 256, 128}, {17, N_NOTES, 256, 256, 128},   // note L1 f/b
    {21, NUM_BEATS,256, 256,  64}, {25, NUM_BEATS,256, 256,  64}, // beat L0 f/b
    {29, NUM_BEATS,128, 128,  64}, {33, NUM_BEATS,128, 128,  64}, // beat L1 f/b
    {37, NUM_MEAS, 128, 128,  64}, {41, NUM_MEAS, 128, 128,  64}, // meas L0 f/b
    {45, NUM_MEAS, 128, 128,  64}, {49, NUM_MEAS, 128, 128,  64}, // meas L1 f/b
    {53, N_NOTES,  523, CMB_PITCH, 64},                           // out LSTM
  };

  // ---- workspace carve-out (256B aligned) ----
  char* ws = (char*)d_ws;
  size_t cur = 0;
  auto carve = [&](size_t bytes) -> char* {
    char* p = ws + cur; cur += (bytes + 255) & ~(size_t)255; return p;
  };
  _Float16* F16A  = (_Float16*)carve((size_t)N_NOTES * CMB_PITCH * 2); // f16 activations
  float*    Gb0   = (float*)carve((size_t)N_NOTES * 512 * 4);          // gates, dir 0
  float*    Gb1   = (float*)carve((size_t)N_NOTES * 512 * 4);          // gates, dir 1
  float*    H1    = (float*)carve((size_t)N_NOTES * 256 * 4);
  float*    HN    = (float*)carve((size_t)N_NOTES * 256 * 4);
  float*    BN    = (float*)carve((size_t)NUM_BEATS * 256 * 4);
  float*    HB1   = (float*)carve((size_t)NUM_BEATS * 128 * 4);
  float*    HBEAT = (float*)carve((size_t)NUM_BEATS * 128 * 4);
  float*    MN    = (float*)carve((size_t)NUM_MEAS * 128 * 4);
  float*    HM1   = (float*)carve((size_t)NUM_MEAS * 128 * 4);
  float*    HMEAS = (float*)carve((size_t)NUM_MEAS * 128 * 4);
  float*    HOUT  = (float*)carve((size_t)N_NOTES * 64 * 4);

  auto cvt = [&](const float* src, _Float16* dst, int rows, int scols, int dpitch) {
    int total = rows * dpitch;
    k_cvt_f16<<<(total + 255) / 256, 256, 0, stream>>>(src, dst, rows, scols, dpitch);
  };

  // ---- pack Wih weights to f16, K-padded ----
  _Float16* Wih16[13];
  for (int u = 0; u < 13; ++u) {
    int G4 = 4 * U[u].H;
    Wih16[u] = (_Float16*)carve((size_t)G4 * U[u].Kp * 2);
    cvt((const float*)d_in[U[u].p], Wih16[u], G4, U[u].K, U[u].Kp);
  }

  auto gemm = [&](int u, const _Float16* A16, float* Gb) {
    int G4 = 4 * U[u].H;
    dim3 gg(G4 / 16, U[u].T / 64);
    k_gemm_wmma<<<gg, 128, 0, stream>>>(A16, Wih16[u], Gb, U[u].Kp, G4);
  };

  // bidirectional pair u (fwd) / u+1 (bwd): both GEMMs, then one 2-block scan
  auto run_pair = [&](int u, const _Float16* A16, float* Hout, int ldh) {
    gemm(u,     A16, Gb0);
    gemm(u + 1, A16, Gb1);
    int H = U[u].H, G4 = 4 * H;
    size_t shm = ((size_t)H * G4 + G4 + H) * 4;
    const float* W0 = (const float*)d_in[U[u].p + 1];
    const float* W1 = (const float*)d_in[U[u + 1].p + 1];
    const float* bi0 = (const float*)d_in[U[u].p + 2];
    const float* bh0 = (const float*)d_in[U[u].p + 3];
    const float* bi1 = (const float*)d_in[U[u + 1].p + 2];
    const float* bh1 = (const float*)d_in[U[u + 1].p + 3];
    if (H == 128)
      k_lstm_scan2<128><<<2, 512, shm, stream>>>(Gb0, Gb1, W0, W1, bi0, bh0, bi1, bh1,
                                                 Hout, U[u].T, ldh);
    else
      k_lstm_scan2<64><<<2, 256, shm, stream>>>(Gb0, Gb1, W0, W1, bi0, bh0, bi1, bh1,
                                                Hout, U[u].T, ldh);
  };

  // ---------------- pipeline ----------------
  // note LSTM layer 0 (input 55 -> pad 64)
  cvt(X, F16A, N_NOTES, 55, 64);
  run_pair(0, F16A, H1, 256);
  // note LSTM layer 1
  cvt(H1, F16A, N_NOTES, 256, 256);
  run_pair(2, F16A, HN, 256);
  // beat pooling (sum of 8) + beat LSTM
  k_pool_sum<<<(NUM_BEATS * 256 + 255) / 256, 256, 0, stream>>>(HN, BN, NUM_BEATS, 8, 256);
  cvt(BN, F16A, NUM_BEATS, 256, 256);
  run_pair(4, F16A, HB1, 128);
  cvt(HB1, F16A, NUM_BEATS, 128, 128);
  run_pair(6, F16A, HBEAT, 128);
  // measure pooling (sum of 4) + measure LSTM
  k_pool_sum<<<(NUM_MEAS * 128 + 255) / 256, 256, 0, stream>>>(HBEAT, MN, NUM_MEAS, 4, 128);
  cvt(MN, F16A, NUM_MEAS, 128, 128);
  run_pair(8, F16A, HM1, 128);
  cvt(HM1, F16A, NUM_MEAS, 128, 128);
  run_pair(10, F16A, HMEAS, 128);
  // combined [N,523] -> f16 padded 544, then output LSTM (unidirectional)
  {
    int total = N_NOTES * CMB_PITCH;
    k_build_combined<<<(total + 255) / 256, 256, 0, stream>>>(HN, HBEAT, HMEAS, Y, F16A, N_NOTES);
  }
  {
    gemm(12, F16A, Gb0);
    int H = 64, G4 = 256;
    size_t shm = ((size_t)H * G4 + G4 + H) * 4;
    const float* W0  = (const float*)d_in[U[12].p + 1];
    const float* bi0 = (const float*)d_in[U[12].p + 2];
    const float* bh0 = (const float*)d_in[U[12].p + 3];
    k_lstm_scan2<64><<<1, 256, shm, stream>>>(Gb0, Gb0, W0, W0, bi0, bh0, bi0, bh0,
                                              HOUT, N_NOTES, 64);
  }
  // final linears + trill masking/sigmoid
  k_final<<<(N_NOTES + 255) / 256, 256, 0, stream>>>(HOUT, FCW, FCB, TRW, TRB, X,
                                                     (float*)d_out, N_NOTES);
}
